// ACEModel_63050119905976
// MI455X (gfx1250) — matrix operand order repaired and checked by reading, not compiled, not run on hardware
//
#include <hip/hip_runtime.h>
#include <stdint.h>

// ---------------- problem constants ----------------
#define B_DIM 4096
#define D_DIM 256
#define H_DIM 512
#define K_MIX 10
#define L_RES 4
#define S_SAMP 10
#define HEAD_FULL 94                 // 3*K + C
#define HEAD_PAD 32                  // 30 live columns padded to 32
#define N_HEAD (D_DIM * HEAD_PAD)    // 8192 packed head columns
#define N_FIN (D_DIM * HEAD_FULL)    // 24064
#define LOG2PI_F 1.8378770664093453f

typedef unsigned short bf16_t;  // raw bf16 storage
typedef __attribute__((ext_vector_type(16))) __bf16 v16bf;
typedef __attribute__((ext_vector_type(8)))  float  v8f;

union FragB16 { v16bf v; uint4 q[2]; };

static __device__ __forceinline__ bf16_t f2bf(float f) {
  uint32_t u = __float_as_uint(f);
  uint32_t r = u + 0x7FFFu + ((u >> 16) & 1u);   // round-to-nearest-even
  return (bf16_t)(r >> 16);
}

static __device__ __forceinline__ v8f v8f_zero() {
  v8f z = {0.f,0.f,0.f,0.f,0.f,0.f,0.f,0.f};
  return z;
}

static __device__ __forceinline__ uint32_t hashmix(uint32_t x) {
  x ^= x >> 16; x *= 0x7FEB352Du;
  x ^= x >> 15; x *= 0x846CA68Bu;
  x ^= x >> 16;
  return x;
}

// ---------------- weight packing ----------------
// W_in / W1 / W2 -> transposed (N-major) bf16: Wt[n][k] = W[k][n].
// Reads coalesced (src[o] with o = k*512+n, n fastest), writes strided.
__global__ void pack_square_k(const float* __restrict__ W_in,
                              const float* __restrict__ W1,
                              const float* __restrict__ W2,
                              bf16_t* __restrict__ dst) {
  int idx = blockIdx.x * blockDim.x + threadIdx.x;
  const int per = H_DIM * H_DIM;
  if (idx >= 9 * per) return;
  int mat = idx / per;
  int o = idx - mat * per;            // o = k*512 + n
  int k = o >> 9;
  int n = o & 511;
  const float* src = (mat == 0) ? W_in
                   : (mat <= 4) ? (W1 + (size_t)(mat - 1) * per)
                                : (W2 + (size_t)(mat - 5) * per);
  dst[(size_t)mat * per + (size_t)n * H_DIM + k] = f2bf(src[o]);
}

// W_fin (512 x 24064): keep only columns d*94 + (0..29), pad to 32, transpose.
// Packed layout: WtF[(d*32+j)*512 + k]. Reads coalesced over packed col.
__global__ void pack_fin_k(const float* __restrict__ W_fin,
                           const float* __restrict__ b_fin,
                           bf16_t* __restrict__ WtF,
                           float* __restrict__ bF) {
  int idx = blockIdx.x * blockDim.x + threadIdx.x;   // 512 * 8192
  if (idx >= H_DIM * N_HEAD) return;
  int c = idx & (N_HEAD - 1);        // packed col = d*32 + j
  int k = idx >> 13;
  int d = c >> 5, j = c & 31;
  float val = (j < 30) ? W_fin[(size_t)k * N_FIN + (size_t)d * HEAD_FULL + j] : 0.f;
  WtF[(size_t)c * H_DIM + k] = f2bf(val);
  if (k == 0) bF[c] = (j < 30) ? b_fin[(size_t)d * HEAD_FULL + j] : 0.f;
}

// Xin[b][j] = j<256 ? x*mask : mask   (bf16, B x 512)
__global__ void build_input_k(const float* __restrict__ x,
                              const float* __restrict__ mask,
                              bf16_t* __restrict__ Xin) {
  int idx = blockIdx.x * blockDim.x + threadIdx.x;   // B*512
  if (idx >= B_DIM * H_DIM) return;
  int b = idx >> 9, j = idx & 511;
  float v = (j < D_DIM) ? x[(size_t)b * D_DIM + j] * mask[(size_t)b * D_DIM + j]
                        : mask[(size_t)b * D_DIM + (j - D_DIM)];
  Xin[idx] = f2bf(v);
}

// ---------------- fragment loaders (CDNA5 WMMA layouts) ----------------
// A (16x32 bf16, row-major): lane L holds row m=L%16,
// K = 8*(L/16)+[0..7] and 16+8*(L/16)+[0..7]  -> two 16B chunks.
static __device__ __forceinline__ void load_a_frag(FragB16& f, const bf16_t* A,
                                                   int row0, int k0, int l16, int lh) {
  const bf16_t* pa = A + (size_t)(row0 + l16) * H_DIM + k0 + 8 * lh;
  f.q[0] = *(const uint4*)pa;
  f.q[1] = *(const uint4*)(pa + 16);
}
// B (32x16 bf16) from transposed weight Bt[n][k]: lane L holds column n=L%16,
// K = 8*(L/16)+[0..7] and 16+8*(L/16)+[0..7]  -> two 16B chunks.
static __device__ __forceinline__ void load_b_frag(FragB16& f, const bf16_t* Bt,
                                                   int col0, int k0, int l16, int lh) {
  const uint4* pb = (const uint4*)(Bt + (size_t)(col0 + l16) * H_DIM + k0 + 8 * lh);
  f.q[0] = pb[0];
  f.q[1] = pb[2];
}

// ---------------- main GEMM with fused epilogue ----------------
// (4096x512)bf16 @ (512x512) + bias [+res]; optionally writes f32 result and/or
// bf16(relu(result)) as the next GEMM's A operand. Block 128x128, wave 32x64.
__global__ __launch_bounds__(256) void gemm512_k(const bf16_t* __restrict__ A,
                                                 const bf16_t* __restrict__ Bt,
                                                 const float* __restrict__ bias,
                                                 const float* __restrict__ res,
                                                 float* __restrict__ outF,
                                                 bf16_t* __restrict__ outB,
                                                 int relu) {
  const int lane = threadIdx.x & 31;
  const int wave = threadIdx.x >> 5;
  const int l16 = lane & 15;
  const int lh  = lane >> 4;
  const int wm = blockIdx.y * 128 + (wave & 3) * 32;
  const int wn = blockIdx.x * 128 + (wave >> 2) * 64;

  v8f acc[2][4];
  #pragma unroll
  for (int i = 0; i < 2; ++i)
    #pragma unroll
    for (int j = 0; j < 4; ++j) acc[i][j] = v8f_zero();

  for (int k0 = 0; k0 < H_DIM; k0 += 32) {
    FragB16 af[2];
    #pragma unroll
    for (int i = 0; i < 2; ++i) load_a_frag(af[i], A, wm + 16 * i, k0, l16, lh);
    FragB16 bf[4];
    #pragma unroll
    for (int j = 0; j < 4; ++j) load_b_frag(bf[j], Bt, wn + 16 * j, k0, l16, lh);
    #pragma unroll
    for (int i = 0; i < 2; ++i)
      #pragma unroll
      for (int j = 0; j < 4; ++j)
        acc[i][j] = __builtin_amdgcn_wmma_f32_16x16x32_bf16(
            false, af[i].v, false, bf[j].v, (short)0, acc[i][j], false, false);
  }

  #pragma unroll
  for (int i = 0; i < 2; ++i) {
    int rowb = wm + 16 * i + 8 * lh;
    #pragma unroll
    for (int j = 0; j < 4; ++j) {
      int col = wn + 16 * j + l16;
      float bs = bias[col];
      #pragma unroll
      for (int v = 0; v < 8; ++v) {
        size_t o = (size_t)(rowb + v) * H_DIM + col;
        float val = acc[i][j][v] + bs;
        if (res) val += res[o];
        if (outF) outF[o] = val;
        if (outB) outB[o] = f2bf(relu ? fmaxf(val, 0.f) : val);
      }
    }
  }
}

// ---------------- head math for one (b, d) pair ----------------
static __device__ __forceinline__ void head_math(const float* __restrict__ hv,
                                                 int b, int d,
                                                 const float* __restrict__ x,
                                                 const float* __restrict__ mask,
                                                 float* __restrict__ out) {
  float lg[K_MIX], mu[K_MIX], sc[K_MIX];
  #pragma unroll
  for (int k = 0; k < K_MIX; ++k) {
    lg[k] = hv[k];
    mu[k] = hv[K_MIX + k];
    float ps = hv[2 * K_MIX + k];
    float sp = (ps > 15.f) ? ps : log1pf(__expf(ps));   // softplus
    sc[k] = sp + 0.001f;
  }
  float q  = 1.f - mask[(size_t)b * D_DIM + d];
  float xu = x[(size_t)b * D_DIM + d] * q;

  float mx = lg[0];
  #pragma unroll
  for (int k = 1; k < K_MIX; ++k) mx = fmaxf(mx, lg[k]);
  float se = 0.f;
  #pragma unroll
  for (int k = 0; k < K_MIX; ++k) se += __expf(lg[k] - mx);
  float logZ = mx + __logf(se);

  float w[K_MIX], t[K_MIX];
  float pmean = 0.f, tmax = -1e30f;
  #pragma unroll
  for (int k = 0; k < K_MIX; ++k) {
    float lw = lg[k] - logZ;
    w[k] = __expf(lw);
    pmean += w[k] * mu[k];
    float z = (xu - mu[k]) / sc[k];
    t[k] = lw - 0.5f * z * z - __logf(sc[k]) - 0.5f * LOG2PI_F;
    tmax = fmaxf(tmax, t[k]);
  }
  float ts = 0.f;
  #pragma unroll
  for (int k = 0; k < K_MIX; ++k) ts += __expf(t[k] - tmax);
  float pll = (tmax + __logf(ts)) * q;
  pmean *= q;

  const size_t BD  = (size_t)B_DIM * D_DIM;
  const size_t BSD = BD * S_SAMP;
  out[(size_t)b * D_DIM + d] = pll;
  out[BD + BSD + (size_t)b * D_DIM + d] = pmean;

  // deterministic counter RNG (replaces JAX threefry; same distribution)
  uint32_t base = (uint32_t)(b * D_DIM + d);
  for (int si = 0; si < S_SAMP; ++si) {
    uint32_t h0 = hashmix(base * 0x9E3779B9u + (uint32_t)si * 0x85EBCA6Bu + 0x1234567u);
    uint32_t h1 = hashmix(h0 ^ 0xC2B2AE35u);
    uint32_t h2 = hashmix(h1 + 0x27D4EB2Fu);
    float u  = (float)(h0 >> 8) * (1.f / 16777216.f);
    float u1 = ((float)(h1 >> 8) + 0.5f) * (1.f / 16777216.f);
    float u2 = (float)(h2 >> 8) * (1.f / 16777216.f);
    int kc = K_MIX - 1;
    float c = 0.f;
    #pragma unroll
    for (int k = 0; k < K_MIX; ++k) { c += w[k]; if (u < c) { kc = k; break; } }
    float eps = sqrtf(fmaxf(-2.f * __logf(u1), 0.f)) * __cosf(6.28318530718f * u2);
    float smp = (mu[kc] + sc[kc] * eps) * q;
    out[BD + ((size_t)b * S_SAMP + si) * D_DIM + d] = smp;
  }
}

// ---------------- fused final GEMM + head ----------------
// Hb(4096x512) @ WtF(512x8192): standard 128x128 block tile (8 waves, 32x64
// wave tiles), epilogue routes the f32 tile through LDS in two 64-row stages
// and computes the full mixture head per (b,d) pair. Block covers 4 d values.
__global__ __launch_bounds__(256) void head_k(const bf16_t* __restrict__ Hb,
                                              const bf16_t* __restrict__ WtF,
                                              const float* __restrict__ bF,
                                              const float* __restrict__ x,
                                              const float* __restrict__ mask,
                                              float* __restrict__ out) {
  const int lane = threadIdx.x & 31;
  const int wave = threadIdx.x >> 5;
  const int l16 = lane & 15, lh = lane >> 4;
  const int bm = blockIdx.y * 128;           // batch rows
  const int bn = blockIdx.x * 128;           // packed head cols (4 d's)
  const int wrow = (wave & 3) * 32;          // local wave row
  const int wcol = (wave >> 2) * 64;         // local wave col

  v8f acc[2][4];
  #pragma unroll
  for (int i = 0; i < 2; ++i)
    #pragma unroll
    for (int j = 0; j < 4; ++j) acc[i][j] = v8f_zero();

  for (int k0 = 0; k0 < H_DIM; k0 += 32) {
    FragB16 af[2];
    #pragma unroll
    for (int i = 0; i < 2; ++i) load_a_frag(af[i], Hb, bm + wrow + 16 * i, k0, l16, lh);
    FragB16 bf[4];
    #pragma unroll
    for (int j = 0; j < 4; ++j) load_b_frag(bf[j], WtF, bn + wcol + 16 * j, k0, l16, lh);
    #pragma unroll
    for (int i = 0; i < 2; ++i)
      #pragma unroll
      for (int j = 0; j < 4; ++j)
        acc[i][j] = __builtin_amdgcn_wmma_f32_16x16x32_bf16(
            false, af[i].v, false, bf[j].v, (short)0, acc[i][j], false, false);
  }

  __shared__ float tile[64][132];   // 33 KB, padded stride (132*8 % 64 == 32)

  #pragma unroll
  for (int half = 0; half < 2; ++half) {
    if ((wrow >> 6) == half) {      // waves owning rows [half*64, half*64+64)
      #pragma unroll
      for (int i = 0; i < 2; ++i) {
        int rbase = (wrow - half * 64) + 16 * i + 8 * lh;
        #pragma unroll
        for (int j = 0; j < 4; ++j) {
          int lc = wcol + 16 * j + l16;
          float bs = bF[bn + lc];
          #pragma unroll
          for (int v = 0; v < 8; ++v)
            tile[rbase + v][lc] = acc[i][j][v] + bs;
        }
      }
    }
    __syncthreads();
    {
      int row = threadIdx.x & 63;   // one (b,d) pair per thread per stage
      int dl  = threadIdx.x >> 6;   // 0..3
      int b = bm + half * 64 + row;
      int d = (bn >> 5) + dl;
      head_math(&tile[row][dl * 32], b, d, x, mask, out);
    }
    __syncthreads();
  }
}

// ---------------- workspace layout ----------------
static constexpr size_t OFF_WTSQ = 0;                              // 9*512*512*2  = 4718592
static constexpr size_t OFF_WTF  = OFF_WTSQ + 9ull*512*512*2;      // 8192*512*2   = 8388608
static constexpr size_t OFF_BF   = OFF_WTF + (size_t)N_HEAD*512*2; // 8192*4       = 32768
static constexpr size_t OFF_XIN  = OFF_BF  + (size_t)N_HEAD*4;     // 4096*512*2   = 4194304
static constexpr size_t OFF_H0   = OFF_XIN + 4096ull*512*2;        // 4096*512*4   = 8388608
static constexpr size_t OFF_H1   = OFF_H0  + 4096ull*512*4;        // 8388608
static constexpr size_t OFF_HB   = OFF_H1  + 4096ull*512*4;        // 4096*512*2   = 4194304
static constexpr size_t OFF_HB2  = OFF_HB  + 4096ull*512*2;        // 4194304
// total ~42.5 MB

extern "C" void kernel_launch(void* const* d_in, const int* in_sizes, int n_in,
                              void* d_out, int out_size, void* d_ws, size_t ws_size,
                              hipStream_t stream) {
  const float* x     = (const float*)d_in[0];
  const float* mask  = (const float*)d_in[1];
  const float* W_in  = (const float*)d_in[2];
  const float* b_in  = (const float*)d_in[3];
  const float* W1    = (const float*)d_in[4];
  const float* b1    = (const float*)d_in[5];
  const float* W2    = (const float*)d_in[6];
  const float* b2    = (const float*)d_in[7];
  const float* W_fin = (const float*)d_in[8];
  const float* b_fin = (const float*)d_in[9];

  char* ws = (char*)d_ws;
  bf16_t* WtIn = (bf16_t*)(ws + OFF_WTSQ);
  bf16_t* Wt1  = WtIn + 1ull * 512 * 512;
  bf16_t* Wt2  = WtIn + 5ull * 512 * 512;
  bf16_t* WtF  = (bf16_t*)(ws + OFF_WTF);
  float*  bF   = (float*) (ws + OFF_BF);
  bf16_t* Xin  = (bf16_t*)(ws + OFF_XIN);
  float*  h0   = (float*) (ws + OFF_H0);
  float*  h1   = (float*) (ws + OFF_H1);
  bf16_t* hb   = (bf16_t*)(ws + OFF_HB);
  bf16_t* hb2  = (bf16_t*)(ws + OFF_HB2);

  {
    int n = 9 * 512 * 512;
    pack_square_k<<<(n + 255) / 256, 256, 0, stream>>>(W_in, W1, W2, WtIn);
  }
  {
    int n = H_DIM * N_HEAD;
    pack_fin_k<<<(n + 255) / 256, 256, 0, stream>>>(W_fin, b_fin, WtF, bF);
  }
  {
    int n = B_DIM * H_DIM;
    build_input_k<<<(n + 255) / 256, 256, 0, stream>>>(x, mask, Xin);
  }

  const dim3 gg(H_DIM / 128, B_DIM / 128);   // (4, 32)

  // h = Xin @ W_in + b_in  -> f32 h0 (residual) + bf16 relu(h) (next A)
  gemm512_k<<<gg, 256, 0, stream>>>(Xin, WtIn, b_in, nullptr, h0, hb, 1);

  float* hcur = h0;
  float* hnext = h1;
  for (int l = 0; l < L_RES; ++l) {
    // r = relu(h) @ W1 + b1 : only bf16 relu(r) is ever needed
    gemm512_k<<<gg, 256, 0, stream>>>(hb, Wt1 + (size_t)l * 512 * 512,
                                      b1 + (size_t)l * 512, nullptr,
                                      nullptr, hb2, 1);
    // h' = relu(r) @ W2 + b2 + h : f32 (residual, unless last layer) +
    // bf16 relu(h') (next A) or bf16 h' (head input) on the last layer
    int last = (l == L_RES - 1);
    gemm512_k<<<gg, 256, 0, stream>>>(hb2, Wt2 + (size_t)l * 512 * 512,
                                      b2 + (size_t)l * 512, hcur,
                                      last ? nullptr : hnext, hb, last ? 0 : 1);
    float* tmp = hcur; hcur = hnext; hnext = tmp;
  }

  // fused final GEMM (4096x512 @ 512x8192) + mixture head + sampling
  head_k<<<dim3(N_HEAD / 128, B_DIM / 128), 256, 0, stream>>>(hb, WtF, bF, x,
                                                              mask, (float*)d_out);
}
